// CylinderSemanticHead_5549097746519
// MI455X (gfx1250) — matrix operand order repaired and checked by reading, not compiled, not run on hardware
//
#include <hip/hip_runtime.h>

// Cylinder3D semantic head: SubMConv3d(128 -> 20, k=3, pad=1), gather-GEMM
// with bf16 WMMA (v_wmma_f32_16x16x32_bf16) on gfx1250 (wave32).
// M-tile = 32 voxels/wave (2 A fragments x 2 N-tiles = 4 accumulators);
// invalid neighbors hit a zeroed pad row -> all gathers branch-free.

typedef __attribute__((ext_vector_type(16))) __bf16       v16bf;
typedef __attribute__((ext_vector_type(8)))  float        v8f;
typedef __attribute__((ext_vector_type(4)))  unsigned int v4u;

#define GD 480
#define GH 360
#define GW 32
#define CIN 128
#define COUT 20
#define NTILES 2
#define KSLICES 4                                   // 128 / 32
#define WSWZ_PER_TAP (KSLICES * NTILES * 32 * 16)   // 4096 bf16 per tap
#define NTAPS 27

// ---------------- prep kernels ----------------

__global__ void k_clear_idx(int* __restrict__ m, int n) {
    int i = blockIdx.x * blockDim.x + threadIdx.x;
    if (i < n) m[i] = -1;
}

__global__ void k_scatter_idx(const int* __restrict__ coords, int* __restrict__ m, int n) {
    int i = blockIdx.x * blockDim.x + threadIdx.x;
    if (i < n) {
        int z = coords[3 * i + 0];
        int y = coords[3 * i + 1];
        int x = coords[3 * i + 2];
        m[(z * GH + y) * GW + x] = i;
    }
}

// Convert features f32->bf16 and zero-fill one extra pad row at index nvox.
__global__ void k_cvt_feat(const float* __restrict__ f, __bf16* __restrict__ o,
                           int n, int ntot) {
    int i = blockIdx.x * blockDim.x + threadIdx.x;
    if (i < ntot) o[i] = (i < n) ? (__bf16)f[i] : (__bf16)0.0f;
}

// Pre-swizzle weights into exact WMMA B-fragment order:
// [tap][kslice][ntile][lane][elem0..15]; elem e at lane L maps to
// K_local = (e<8 ? e : e+8) + 8*(L/16), N = (L%16) + 16*ntile   (32x16 bf16 B layout).
__global__ void k_wswz(const float* __restrict__ w, __bf16* __restrict__ o) {
    int i = blockIdx.x * blockDim.x + threadIdx.x;
    if (i >= NTAPS * WSWZ_PER_TAP) return;
    int e     = i & 15;
    int lane  = (i >> 4) & 31;
    int ntile = (i >> 9) & 1;
    int ks    = (i >> 10) & 3;
    int tap   = i >> 12;
    int half  = lane >> 4;
    int n     = ntile * 16 + (lane & 15);
    int K     = ks * 32 + (e < 8 ? e : e + 8) + 8 * half;
    float val = (n < COUT) ? w[(tap * CIN + K) * COUT + n] : 0.0f;
    o[i] = (__bf16)val;
}

// ---------------- main gather-GEMM kernel ----------------

__global__ void __launch_bounds__(256)
k_subm_conv(const __bf16* __restrict__ feat,    // [nvox+1][128] bf16 (last row = 0)
            const __bf16* __restrict__ wswz,    // swizzled B fragments
            const float*  __restrict__ bias,    // [20]
            const int*    __restrict__ coords,  // [nvox][3]
            const int*    __restrict__ idxmap,  // [GD*GH*GW]
            float*        __restrict__ out,     // [nvox][20]
            int nvox) {
    const int lane  = threadIdx.x & 31;
    const int wave  = blockIdx.x * (blockDim.x >> 5) + (threadIdx.x >> 5);
    const int tiles = (nvox + 31) >> 5;             // 32 voxels per wave
    if (wave >= tiles) return;                      // wave-uniform, EXEC stays full

    const int half  = lane >> 4;
    const int row   = lane & 15;                    // A-matrix row owned by lane
    const int vbase = wave << 5;
    const int v0    = vbase + row;                  // A-tile 0 voxel
    const int v1    = vbase + 16 + row;             // A-tile 1 voxel
    const bool vok0 = v0 < nvox;
    const bool vok1 = v1 < nvox;

    // Clamped, unconditional coordinate loads (no EXEC-masked branches).
    const int vc0 = vok0 ? v0 : 0;
    const int vc1 = vok1 ? v1 : 0;
    const int cz0 = coords[3 * vc0 + 0], cy0 = coords[3 * vc0 + 1], cx0 = coords[3 * vc0 + 2];
    const int cz1 = coords[3 * vc1 + 0], cy1 = coords[3 * vc1 + 1], cx1 = coords[3 * vc1 + 2];

    // Bias folded into accumulators (C/D layout: N = lane%16 + 16*ntile).
    const float b0 = (row < COUT) ? bias[row] : 0.0f;
    const float b1 = ((16 + row) < COUT) ? bias[16 + row] : 0.0f;
    v8f acc00, acc01, acc10, acc11;   // acc[Atile][ntile]
#pragma unroll
    for (int r = 0; r < 8; ++r) {
        acc00[r] = b0; acc01[r] = b1;
        acc10[r] = b0; acc11[r] = b1;
    }

    // 16-bit A 16x32 layout: lane holds K in [8*half, +8) and [16+8*half, +8).
    const int kb1 = 8 * half;
    const int kb2 = 16 + 8 * half;

    int tap = 0;
#pragma unroll 1
    for (int dz = -1; dz <= 1; ++dz)
#pragma unroll 1
    for (int dy = -1; dy <= 1; ++dy)
#pragma unroll 1
    for (int dx = -1; dx <= 1; ++dx, ++tap) {
        // Branch-free neighbor resolution: clamped idxmap address, pad-row select.
        const int nz0 = cz0 + dz, ny0 = cy0 + dy, nx0 = cx0 + dx;
        const int nz1 = cz1 + dz, ny1 = cy1 + dy, nx1 = cx1 + dx;
        const bool inb0 = vok0 && (unsigned)nz0 < GD && (unsigned)ny0 < GH && (unsigned)nx0 < GW;
        const bool inb1 = vok1 && (unsigned)nz1 < GD && (unsigned)ny1 < GH && (unsigned)nx1 < GW;
        const int lin0 = inb0 ? (nz0 * GH + ny0) * GW + nx0 : 0;
        const int lin1 = inb1 ? (nz1 * GH + ny1) * GW + nx1 : 0;
        const int ni0 = idxmap[lin0];
        const int ni1 = idxmap[lin1];
        const int r0 = (inb0 && ni0 >= 0) ? ni0 : nvox;   // nvox = zero pad row
        const int r1 = (inb1 && ni1 >= 0) ? ni1 : nvox;
        const __bf16* arow0 = feat + (size_t)r0 * CIN;
        const __bf16* arow1 = feat + (size_t)r1 * CIN;
        const __bf16* btap  = wswz + tap * WSWZ_PER_TAP;

#pragma unroll
        for (int ks = 0; ks < KSLICES; ++ks) {
            // A fragments: unconditional dual-b128 gathers (pad row handles invalid)
            union { v4u u[2]; v16bf v; } A0, A1;
            A0.u[0] = *(const v4u*)(arow0 + ks * 32 + kb1);
            A0.u[1] = *(const v4u*)(arow0 + ks * 32 + kb2);
            A1.u[0] = *(const v4u*)(arow1 + ks * 32 + kb1);
            A1.u[1] = *(const v4u*)(arow1 + ks * 32 + kb2);

            // B fragments: contiguous 32B per lane (pre-swizzled, L2-resident),
            // each reused by 2 WMMAs (both A tiles).
            const __bf16* bp0 = btap + (ks * NTILES + 0) * 512 + lane * 16;
            const __bf16* bp1 = btap + (ks * NTILES + 1) * 512 + lane * 16;
            union { v4u u[2]; v16bf v; } B0, B1;
            B0.u[0] = *(const v4u*)(bp0);
            B0.u[1] = *(const v4u*)(bp0 + 8);
            B1.u[0] = *(const v4u*)(bp1);
            B1.u[1] = *(const v4u*)(bp1 + 8);

            acc00 = __builtin_amdgcn_wmma_f32_16x16x32_bf16(
                false, A0.v, false, B0.v, (short)0, acc00, false, false);
            acc01 = __builtin_amdgcn_wmma_f32_16x16x32_bf16(
                false, A0.v, false, B1.v, (short)0, acc01, false, false);
            acc10 = __builtin_amdgcn_wmma_f32_16x16x32_bf16(
                false, A1.v, false, B0.v, (short)0, acc10, false, false);
            acc11 = __builtin_amdgcn_wmma_f32_16x16x32_bf16(
                false, A1.v, false, B1.v, (short)0, acc11, false, false);
        }
    }

    // Store: C/D layout -> M = base + r + 8*half, N = row (+16 for ntile 1)
#pragma unroll
    for (int r = 0; r < 8; ++r) {
        const int m0 = vbase + r + 8 * half;        // A-tile 0 rows
        const int m1 = m0 + 16;                     // A-tile 1 rows
        if (m0 < nvox) {
            if (row < COUT)      out[m0 * COUT + row]      = acc00[r];
            if (16 + row < COUT) out[m0 * COUT + 16 + row] = acc01[r];
        }
        if (m1 < nvox) {
            if (row < COUT)      out[m1 * COUT + row]      = acc10[r];
            if (16 + row < COUT) out[m1 * COUT + 16 + row] = acc11[r];
        }
    }
}

// ---------------- host entry ----------------

extern "C" void kernel_launch(void* const* d_in, const int* in_sizes, int n_in,
                              void* d_out, int out_size, void* d_ws, size_t ws_size,
                              hipStream_t stream) {
    const float* features = (const float*)d_in[0];   // [N,128] f32
    const float* weight   = (const float*)d_in[1];   // [27,128,20] f32
    const float* bias     = (const float*)d_in[2];   // [20] f32
    const int*   coords   = (const int*)d_in[3];     // [N,3] i32
    const int nvox = in_sizes[0] / CIN;

    // Workspace partition (aligned to 256B):
    char* ws = (char*)d_ws;
    size_t off = 0;
    int* idxmap = (int*)(ws + off);
    off += (size_t)GD * GH * GW * sizeof(int);
    off = (off + 255) & ~(size_t)255;
    __bf16* featb = (__bf16*)(ws + off);
    off += (size_t)(nvox + 1) * CIN * sizeof(__bf16);   // +1 zero pad row
    off = (off + 255) & ~(size_t)255;
    __bf16* wswz = (__bf16*)(ws + off);

    const int dhw = GD * GH * GW;
    k_clear_idx<<<(dhw + 255) / 256, 256, 0, stream>>>(idxmap, dhw);
    k_scatter_idx<<<(nvox + 255) / 256, 256, 0, stream>>>(coords, idxmap, nvox);

    const int fe   = nvox * CIN;
    const int ftot = (nvox + 1) * CIN;
    k_cvt_feat<<<(ftot + 255) / 256, 256, 0, stream>>>(features, featb, fe, ftot);

    const int we = NTAPS * WSWZ_PER_TAP;
    k_wswz<<<(we + 255) / 256, 256, 0, stream>>>(weight, wswz);

    const int tiles  = (nvox + 31) / 32;   // one wave32 per 32-voxel tile
    const int blocks = (tiles + 7) / 8;    // 8 waves per 256-thread block
    k_subm_conv<<<blocks, 256, 0, stream>>>(featb, wswz, bias, coords, idxmap,
                                            (float*)d_out, nvox);
}
